// LSTMDriftCompensator_42691974922241
// MI455X (gfx1250) — compile-verified, matrix operand order checked by reading, not run on hardware
//
#include <hip/hip_runtime.h>

#define TSEQ 512
#define BATCH 1024
#define DIN 12
#define HID 128

typedef _Float16 v16h __attribute__((ext_vector_type(16)));
typedef _Float16 h8   __attribute__((ext_vector_type(8)));
typedef float    v8f  __attribute__((ext_vector_type(8)));
typedef unsigned int v4u __attribute__((ext_vector_type(4)));
typedef int      v8i  __attribute__((ext_vector_type(8)));
typedef int      v4i  __attribute__((ext_vector_type(4)));

#if __has_builtin(__builtin_amdgcn_tensor_load_to_lds) && __has_builtin(__builtin_amdgcn_s_wait_tensorcnt)
#define HAS_TDM 1
#else
#define HAS_TDM 0
#endif

// ---- WMMA 16x16x32 f16 fragment layout helpers (wave32) -------------------
// A (16x32, MxK): lane<16 -> M=lane, K in {0..7,16..23}; lane>=16 -> M=lane-16,
// K in {8..15,24..31}. Element idx i (0..15): base=(i<8)?i:i+8, +8 if hi lane.
// B (32x16, KxN) mirrored: N across lanes, same K striping.
__device__ __forceinline__ int fragK(int lane, int idx) {
  int base = (idx < 8) ? idx : idx + 8;
  return base + ((lane >> 4) << 3);
}
__device__ __forceinline__ void fragPos(int klocal, int& laneHi, int& idx) {
  laneHi = (klocal >> 3) & 1;
  int t = klocal - (laneHi << 3);
  idx = (t < 8) ? t : t - 8;
}
__device__ __forceinline__ v16h frag_ld(const _Float16* p) {
  const h8* q = (const h8*)p;
  h8 lo = q[0], hi = q[1];
  return __builtin_shufflevector(lo, hi, 0,1,2,3,4,5,6,7,8,9,10,11,12,13,14,15);
}
__device__ __forceinline__ v16h frag_ld2(const _Float16* plo, const _Float16* phi) {
  h8 lo = *(const h8*)plo;
  h8 hi = *(const h8*)phi;
  return __builtin_shufflevector(lo, hi, 0,1,2,3,4,5,6,7,8,9,10,11,12,13,14,15);
}
__device__ __forceinline__ v8f wmma16(v16h a, v16h b, v8f c) {
  return __builtin_amdgcn_wmma_f32_16x16x32_f16(false, a, false, b, (short)0, c,
                                                false, false);
}
__device__ __forceinline__ float sigm(float x) { return 1.0f / (1.0f + __expf(-x)); }
__device__ __forceinline__ float tanh_f(float x) {
  x = fminf(fmaxf(x, -15.0f), 15.0f);
  float e = __expf(2.0f * x);
  return (e - 1.0f) / (e + 1.0f);
}

__device__ __forceinline__ unsigned lds_off(const void* p) {
  return (unsigned)(size_t)(__attribute__((address_space(3))) const void*)p;
}

// ---- TDM: 1-D global->LDS DMA of `bytes` (multiple of 8, <= 65535*8) ------
__device__ __forceinline__ void tdm_load_1d(const void* gsrc, unsigned ldsOff,
                                            unsigned bytes) {
#if HAS_TDM
  unsigned n8 = bytes >> 3;  // 8-byte elements
  unsigned long long ga = (unsigned long long)gsrc;
  v4u g0;
  g0[0] = 1u;                                    // count=1 (valid user D#)
  g0[1] = ldsOff;                                // lds_addr (bytes)
  g0[2] = (unsigned)(ga & 0xffffffffu);          // global_addr[31:0]
  g0[3] = (unsigned)((ga >> 32) & 0x1ffffffu) | (2u << 30);  // addr[56:32]|type=2
  v8i g1;
  g1[0] = (3 << 16);                             // data_size = 8B
  g1[1] = (int)((n8 & 0xffffu) << 16);           // tensor_dim0[15:0]
  g1[2] = (int)(((n8 >> 16) & 0xffffu) | (1u << 16)); // dim0[31:16] | dim1=1 lo
  g1[3] = (int)((n8 & 0xffffu) << 16);           // dim1 hi=0 | tile_dim0=n8
  g1[4] = 1;                                     // tile_dim1=1, tile_dim2=0
  g1[5] = (int)n8;                               // tensor_dim0_stride[31:0]
  g1[6] = (int)((n8 & 0xffffu) << 16);           // stride0 hi=0 | stride1 lo16
  g1[7] = (int)((n8 >> 16) & 0xffffu);           // stride1[47:16]
  v4i z4 = {0, 0, 0, 0};
  v8i z8 = {0, 0, 0, 0, 0, 0, 0, 0};
  __builtin_amdgcn_tensor_load_to_lds(g0, g1, z4, z4, z8, 0);
  __builtin_amdgcn_s_wait_tensorcnt(0);
#else
  (void)gsrc; (void)ldsOff; (void)bytes;
#endif
}

// ---- workspace layout (in _Float16 elements) ------------------------------
// [0      , 65536 ) Whh0 frags   (kb*32+nt)*512
// [65536  , 81920 ) Wih0x frags  (K block: 0..11=x, 12..31=pad)
// [81920  , 147456) Wih1 frags
// [147456 , 212992) Whh1 frags
// [212992 , ...   ) h1 staging: [bt][t][m(16)][HID] f16
#define WS_WIH1  81920
#define WS_H1    212992

// ===========================================================================
__global__ void pack_weights(const float* __restrict__ Whh0,
                             const float* __restrict__ Wih0,
                             const float* __restrict__ Wih1,
                             const float* __restrict__ Whh1,
                             _Float16* __restrict__ wsW) {
  int e = blockIdx.x * 256 + threadIdx.x;  // 416 frags * 512 elems
  if (e >= 416 * 512) return;
  int frag = e >> 9, within = e & 511;
  int lane = within >> 4, idx = within & 15;
  const float* W; int Kvalid, ld, kb, nt;
  if (frag < 128)      { W = Whh0; Kvalid = 128; ld = 128; int lf = frag;       kb = lf >> 5; nt = lf & 31; }
  else if (frag < 160) { W = Wih0; Kvalid = 12;  ld = 12;  int lf = frag - 128; kb = 0;       nt = lf;      }
  else if (frag < 288) { W = Wih1; Kvalid = 128; ld = 128; int lf = frag - 160; kb = lf >> 5; nt = lf & 31; }
  else                 { W = Whh1; Kvalid = 128; ld = 128; int lf = frag - 288; kb = lf >> 5; nt = lf & 31; }
  int K = kb * 32 + fragK(lane, idx);
  int N = nt * 16 + (lane & 15);
  float v = (K < Kvalid) ? W[N * ld + K] : 0.0f;  // B[k][n] = W[n][k]
  wsW[e] = (_Float16)v;
}

// ===========================================================================
// Layer 0: persistent per 16-batch-row tile.  gates(16x512) = [h | x-pad] x W
__global__ void __launch_bounds__(256)
lstm_layer0(const float* __restrict__ x, const float* __restrict__ bih,
            const float* __restrict__ bhh, const _Float16* __restrict__ wsW,
            _Float16* __restrict__ h1s) {
  extern __shared__ _Float16 sm0[];
  _Float16* sWhh = sm0;                  // 65536
  _Float16* sWx  = sm0 + 65536;          // 16384
  _Float16* sH   = sm0 + 81920;          // 2 * 2048 (double buffered frag layout)
  _Float16* sX   = sm0 + 86016;          // 2 * 512
  const int tid = threadIdx.x;
  const int wave = tid >> 5, lane = tid & 31;
  const int bt = blockIdx.x;

#if HAS_TDM
  if (wave == 0) tdm_load_1d(wsW, lds_off(sm0), 81920u * 2u);
#else
  {
    const uint4* s = (const uint4*)wsW;
    uint4* d = (uint4*)sm0;
    for (int i = tid; i < 10240; i += 256) d[i] = s[i];
  }
#endif
  { // zero h and x buffers (h0 = 0, x pad lanes = 0)
    unsigned int* z = (unsigned int*)(sm0 + 81920);
    for (int i = tid; i < 2560; i += 256) z[i] = 0u;
  }
  __syncthreads();

  // hoist all loop-invariant B fragments into registers (weights never change)
  v16h bX[4], bH[4][4];
#pragma unroll
  for (int g = 0; g < 4; ++g) {
    const int nt = wave + 8 * g;
    bX[g] = frag_ld(sWx + (nt << 9) + (lane << 4));
#pragma unroll
    for (int kb = 0; kb < 4; ++kb)
      bH[g][kb] = frag_ld(sWhh + (((kb << 5) + nt) << 9) + (lane << 4));
  }
  if (tid < 192) { // x for t=0 into buffer 0 (A-fragment layout)
    int m = tid / 12, d = tid - 12 * m;
    float xv = x[(size_t)(bt * 16 + m) * (TSEQ * DIN) + d];
    int lh, idx; fragPos(d, lh, idx);
    sX[((m + (lh << 4)) << 4) + idx] = (_Float16)xv;
  }
  float bsum[4];
#pragma unroll
  for (int g = 0; g < 4; ++g) {
    int n = ((wave + 8 * g) << 4) + (lane & 15);
    bsum[g] = bih[n] + bhh[n];
  }
  float cst[8];
#pragma unroll
  for (int i = 0; i < 8; ++i) cst[i] = 0.0f;
  int klocal = ((wave & 1) << 4) | (lane & 15);
  int wLH, wIdx; fragPos(klocal, wLH, wIdx);
  const int wKb = wave >> 1;
  __syncthreads();

  for (int t = 0; t < TSEQ; ++t) {
    const int p = t & 1;
    v16h aX = frag_ld(sX + (p << 9) + (lane << 4));
    v16h aH[4];
#pragma unroll
    for (int kb = 0; kb < 4; ++kb)
      aH[kb] = frag_ld(sH + (p << 11) + (kb << 9) + (lane << 4));
    v8f acc[4];
#pragma unroll
    for (int g = 0; g < 4; ++g) {
      v8f a;
#pragma unroll
      for (int i = 0; i < 8; ++i) a[i] = bsum[g];  // bias folded into C
#pragma unroll
      for (int kb = 0; kb < 4; ++kb) a = wmma16(aH[kb], bH[g][kb], a);
      a = wmma16(aX, bX[g], a);
      acc[g] = a;
    }
    if (t + 1 < TSEQ && tid < 192) { // stage next x into other buffer
      int m = tid / 12, d = tid - 12 * m;
      float xv = x[(size_t)(bt * 16 + m) * (TSEQ * DIN) + (size_t)(t + 1) * DIN + d];
      int lh, idx; fragPos(d, lh, idx);
      sX[((1 - p) << 9) + ((m + (lh << 4)) << 4) + idx] = (_Float16)xv;
    }
    const size_t gbase = ((size_t)bt * TSEQ + t) * (16 * HID) + (wave << 4) + (lane & 15);
#pragma unroll
    for (int v = 0; v < 8; ++v) {
      float iv = sigm(acc[0][v]);
      float fv = sigm(acc[1][v]);
      float gv = tanh_f(acc[2][v]);
      float ov = sigm(acc[3][v]);
      float c = fv * cst[v] + iv * gv;
      cst[v] = c;
      float hv = ov * tanh_f(c);
      int M = v + ((lane >> 4) << 3);
      sH[((1 - p) << 11) + (wKb << 9) + ((M + (wLH << 4)) << 4) + wIdx] = (_Float16)hv;
      h1s[gbase + (size_t)M * HID] = (_Float16)hv;  // stage for layer 1
    }
    __syncthreads();
  }
}

// ===========================================================================
// Layer 1 + head: gates = [h1_t | h2_{t-1}] x [Wih1 ; Whh1]  (xp1 GEMM fused)
__global__ void __launch_bounds__(256)
lstm_layer1(const _Float16* __restrict__ h1s, const float* __restrict__ bih,
            const float* __restrict__ bhh, const _Float16* __restrict__ wsW,
            const float* __restrict__ W1, const float* __restrict__ b1,
            const float* __restrict__ W2, const float* __restrict__ b2,
            const float* __restrict__ W3, const float* __restrict__ b3,
            float* __restrict__ out) {
  extern __shared__ _Float16 sm1[];
  _Float16* sWih = sm1;                  // 65536
  _Float16* sWhh = sm1 + 65536;          // 65536
  _Float16* sH   = sm1 + 131072;         // 2 * 2048
  float* sLast = (float*)(sm1 + 135168); // 16*128
  float* sZ1   = sLast + 2048;           // 16*64
  float* sZ2   = sZ1 + 1024;             // 16*32
  const int tid = threadIdx.x;
  const int wave = tid >> 5, lane = tid & 31;
  const int bt = blockIdx.x;

#if HAS_TDM
  if (wave == 0) tdm_load_1d(wsW, lds_off(sm1), 131072u * 2u);
#else
  {
    const uint4* s = (const uint4*)wsW;
    uint4* d = (uint4*)sm1;
    for (int i = tid; i < 16384; i += 256) d[i] = s[i];
  }
#endif
  {
    unsigned int* z = (unsigned int*)(sm1 + 131072);
    for (int i = tid; i < 2048; i += 256) z[i] = 0u;  // h2_0 = 0
  }
  __syncthreads();

  // hoist recurrence weights (Whh1) into registers for the serial chain
  v16h bHh[4][4];
#pragma unroll
  for (int g = 0; g < 4; ++g) {
    const int nt = wave + 8 * g;
#pragma unroll
    for (int kb = 0; kb < 4; ++kb)
      bHh[g][kb] = frag_ld(sWhh + (((kb << 5) + nt) << 9) + (lane << 4));
  }
  float bsum[4];
#pragma unroll
  for (int g = 0; g < 4; ++g) {
    int n = ((wave + 8 * g) << 4) + (lane & 15);
    bsum[g] = bih[n] + bhh[n];
  }
  float cst[8];
#pragma unroll
  for (int i = 0; i < 8; ++i) cst[i] = 0.0f;
  int klocal = ((wave & 1) << 4) | (lane & 15);
  int wLH, wIdx; fragPos(klocal, wLH, wIdx);
  const int wKb = wave >> 1;
  const int mRow = lane & 15;
  const int hOff = (lane >> 4) << 3;
  __syncthreads();

  for (int t = 0; t < TSEQ; ++t) {
    const int p = t & 1;
    // issue global A-fragment loads for h1_t first (latency hidden by Whh chain)
    const _Float16* row = h1s + ((size_t)bt * TSEQ + t) * (16 * HID) + (size_t)mRow * HID;
    v16h aP[4];
#pragma unroll
    for (int kb = 0; kb < 4; ++kb)
      aP[kb] = frag_ld2(row + kb * 32 + hOff, row + kb * 32 + 16 + hOff);
    if (t + 1 < TSEQ)
      __builtin_prefetch((const void*)(h1s + ((size_t)bt * TSEQ + t + 1) * (16 * HID) + lane * 64), 0, 2);
    v16h aH[4];
#pragma unroll
    for (int kb = 0; kb < 4; ++kb)
      aH[kb] = frag_ld(sH + (p << 11) + (kb << 9) + (lane << 4));
    v8f acc[4];
#pragma unroll
    for (int g = 0; g < 4; ++g) {  // register-B recurrence chain first
      v8f a;
#pragma unroll
      for (int i = 0; i < 8; ++i) a[i] = bsum[g];
#pragma unroll
      for (int kb = 0; kb < 4; ++kb) a = wmma16(aH[kb], bHh[g][kb], a);
      acc[g] = a;
    }
#pragma unroll
    for (int g = 0; g < 4; ++g) {  // then the h1 x Wih1 chain (LDS B)
      const int nt = wave + 8 * g;
      v8f a = acc[g];
#pragma unroll
      for (int kb = 0; kb < 4; ++kb)
        a = wmma16(aP[kb], frag_ld(sWih + (((kb << 5) + nt) << 9) + (lane << 4)), a);
      acc[g] = a;
    }
#pragma unroll
    for (int v = 0; v < 8; ++v) {
      float iv = sigm(acc[0][v]);
      float fv = sigm(acc[1][v]);
      float gv = tanh_f(acc[2][v]);
      float ov = sigm(acc[3][v]);
      float c = fv * cst[v] + iv * gv;
      cst[v] = c;
      float hv = ov * tanh_f(c);
      int M = v + ((lane >> 4) << 3);
      sH[((1 - p) << 11) + (wKb << 9) + ((M + (wLH << 4)) << 4) + wIdx] = (_Float16)hv;
      if (t == TSEQ - 1) sLast[M * HID + (wave << 4) + (lane & 15)] = hv;
    }
    __syncthreads();
  }

  // ---- MLP head on last h2 (16 rows) ----
  for (int i0 = tid; i0 < 16 * 64; i0 += 256) {
    int m = i0 >> 6, o = i0 & 63;
    float a = b1[o];
    for (int j = 0; j < 128; ++j) a += sLast[m * 128 + j] * W1[o * 128 + j];
    sZ1[m * 64 + o] = fmaxf(a, 0.0f);
  }
  __syncthreads();
  for (int i0 = tid; i0 < 16 * 32; i0 += 256) {
    int m = i0 >> 5, o = i0 & 31;
    float a = b2[o];
    for (int j = 0; j < 64; ++j) a += sZ1[m * 64 + j] * W2[o * 64 + j];
    sZ2[m * 32 + o] = fmaxf(a, 0.0f);
  }
  __syncthreads();
  if (tid < 32) {
    int m = tid >> 1, o = tid & 1;
    float a = b3[o];
    for (int j = 0; j < 32; ++j) a += sZ2[m * 32 + j] * W3[o * 32 + j];
    out[(size_t)(bt * 16 + m) * 2 + o] = a;
  }
}

// ===========================================================================
extern "C" void kernel_launch(void* const* d_in, const int* in_sizes, int n_in,
                              void* d_out, int out_size, void* d_ws, size_t ws_size,
                              hipStream_t stream) {
  (void)in_sizes; (void)n_in; (void)out_size; (void)ws_size;
  const float* x    = (const float*)d_in[0];
  const float* Wih0 = (const float*)d_in[1];
  const float* Whh0 = (const float*)d_in[2];
  const float* bih0 = (const float*)d_in[3];
  const float* bhh0 = (const float*)d_in[4];
  const float* Wih1 = (const float*)d_in[5];
  const float* Whh1 = (const float*)d_in[6];
  const float* bih1 = (const float*)d_in[7];
  const float* bhh1 = (const float*)d_in[8];
  const float* W1 = (const float*)d_in[9];
  const float* b1 = (const float*)d_in[10];
  const float* W2 = (const float*)d_in[11];
  const float* b2 = (const float*)d_in[12];
  const float* W3 = (const float*)d_in[13];
  const float* b3 = (const float*)d_in[14];
  float* out = (float*)d_out;
  _Float16* wsW = (_Float16*)d_ws;
  _Float16* h1s = wsW + WS_H1;

  pack_weights<<<832, 256, 0, stream>>>(Whh0, Wih0, Wih1, Whh1, wsW);

  const size_t smem0 = 87040 * sizeof(_Float16);                 // 174080 B
  (void)hipFuncSetAttribute((const void*)lstm_layer0,
                            hipFuncAttributeMaxDynamicSharedMemorySize, (int)smem0);
  lstm_layer0<<<BATCH / 16, 256, smem0, stream>>>(x, bih0, bhh0, wsW, h1s);

  const size_t smem1 = 135168 * sizeof(_Float16) + 3584 * sizeof(float); // 284672 B
  (void)hipFuncSetAttribute((const void*)lstm_layer1,
                            hipFuncAttributeMaxDynamicSharedMemorySize, (int)smem1);
  lstm_layer1<<<BATCH / 16, 256, smem1, stream>>>(h1s, bih1, bhh1, wsW + WS_WIH1,
                                                  W1, b1, W2, b2, W3, b3, out);
}
